// MoEBlock_8083128451224
// MI455X (gfx1250) — matrix operand order repaired and checked by reading, not compiled, not run on hardware
//
#include <hip/hip_runtime.h>

// ---------- types ----------
typedef __bf16 bf16_t;
typedef bf16_t v16bf __attribute__((ext_vector_type(16)));
typedef float  v8f   __attribute__((ext_vector_type(8)));

union Frag {
  v16bf v;
  uint4 q[2];
  bf16_t s[16];
};

#define B_   4
#define T_   1024
#define D_   1024
#define H_   16
#define HD_  64
#define E_   8
#define KTOP 2
#define N_   (B_*T_)
#define CAP_ 1280
#define SCALE_ 0.125f
#define NEG_INF (-3.0e38f)

static __device__ __forceinline__ v8f wmma_bf16(const v16bf& a, const v16bf& b, const v8f& c) {
  return __builtin_amdgcn_wmma_f32_16x16x32_bf16(false, a, false, b, (short)0, c, false, false);
}

// ---------- layernorm: f32 row -> bf16 (and optional f32) ----------
__global__ __launch_bounds__(256) void ln_kernel(const float* __restrict__ x,
                                                 const float* __restrict__ w,
                                                 const float* __restrict__ bsh,
                                                 bf16_t* __restrict__ yb,
                                                 float* __restrict__ yf) {
  int row = blockIdx.x, t = threadIdx.x;
  const float* xr = x + (size_t)row * D_;
  float4 v = *(const float4*)(xr + t * 4);
  float s  = v.x + v.y + v.z + v.w;
  float s2 = v.x*v.x + v.y*v.y + v.z*v.z + v.w*v.w;
  __shared__ float rs[256], rq[256];
  rs[t] = s; rq[t] = s2; __syncthreads();
  for (int st = 128; st > 0; st >>= 1) {
    if (t < st) { rs[t] += rs[t + st]; rq[t] += rq[t + st]; }
    __syncthreads();
  }
  float mu  = rs[0] * (1.0f / D_);
  float var = rq[0] * (1.0f / D_) - mu * mu;
  float r   = rsqrtf(var + 1e-5f);
  int d0 = t * 4;
  float4 wv = *(const float4*)(w + d0);
  float4 bv = *(const float4*)(bsh + d0);
  float o0 = (v.x - mu) * r * wv.x + bv.x;
  float o1 = (v.y - mu) * r * wv.y + bv.y;
  float o2 = (v.z - mu) * r * wv.z + bv.z;
  float o3 = (v.w - mu) * r * wv.w + bv.w;
  size_t base = (size_t)row * D_ + d0;
  yb[base + 0] = (bf16_t)o0; yb[base + 1] = (bf16_t)o1;
  yb[base + 2] = (bf16_t)o2; yb[base + 3] = (bf16_t)o3;
  if (yf) { yf[base+0]=o0; yf[base+1]=o1; yf[base+2]=o2; yf[base+3]=o3; }
}

// ---------- W[No,Kd] f32 -> Wt[Kd,No] bf16 (tiled transpose+convert) ----------
__global__ __launch_bounds__(256) void transpack_kernel(const float* __restrict__ W,
                                                        bf16_t* __restrict__ Wt,
                                                        int No, int Kd) {
  __shared__ float tile[32][33];
  int k0 = blockIdx.x * 32, n0 = blockIdx.y * 32;
  int tx = threadIdx.x, ty = threadIdx.y; // 32 x 8
  for (int r = ty; r < 32; r += 8)
    tile[r][tx] = W[(size_t)(n0 + r) * Kd + k0 + tx];
  __syncthreads();
  for (int r = ty; r < 32; r += 8)
    Wt[(size_t)(k0 + r) * No + n0 + tx] = (bf16_t)tile[tx][r];
}

// ---------- generic WMMA GEMM: C[M,N] = A[M,K](bf16) x B[K,N] ----------
// BF32: B matrix is f32 (converted to bf16 on the way into LDS).
// OUTM: 0=f32, 1=bf16, 2=f32+resid.
// LDS-staged, double-buffered: each global byte is read once per workgroup;
// next k-step's global loads issue before the WMMA block (latency hidden),
// one barrier per k-step.
template <int BF32, int OUTM, int GELU>
__global__ __launch_bounds__(256) void gemm_wmma(const bf16_t* __restrict__ A, int lda,
                                                 const void* __restrict__ Bp, int ldb,
                                                 void* __restrict__ C, int ldc,
                                                 const float* __restrict__ resid) {
  __shared__ __align__(16) bf16_t ldsA[2][128 * 32]; // 128 rows x 32 k  (8KB x2)
  __shared__ __align__(16) bf16_t ldsB[2][32 * 128]; // 32 k x 128 cols  (8KB x2)

  int lane = threadIdx.x;
  int w    = threadIdx.y;
  int tid  = w * 32 + lane;
  int waveM = (w >> 1) * 32;
  int waveN = (w & 1) * 64;
  int blockM = blockIdx.y * 128;
  int blockN = blockIdx.x * 128;
  int l15 = lane & 15;
  int hi  = lane >> 4;
  int sel = hi * 8;
  const int Kd = lda; // all A matrices are dense row-major: lda == K

  // cooperative staging coordinates (256 threads, 16 elems each per tile)
  int arow = tid >> 1;          // 0..127
  int acol = (tid & 1) * 16;    // 0 / 16
  int brow = tid >> 3;          // 0..31
  int bcol = (tid & 7) * 16;    // 0..112

  const bf16_t* aSrc  = A + (size_t)(blockM + arow) * lda + acol;
  const float*  bSrcF = (const float*)Bp  + (size_t)brow * ldb + blockN + bcol;
  const bf16_t* bSrcH = (const bf16_t*)Bp + (size_t)brow * ldb + blockN + bcol;

  v8f acc[2][4] = {};

  uint4  stA0, stA1;   // 16 bf16 of A
  float4 stBf[4];      // 16 f32 of B (BF32 path)
  uint4  stB0, stB1;   // 16 bf16 of B

  auto gload = [&](int k0) {
    const bf16_t* ap = aSrc + k0;
    stA0 = *(const uint4*)ap;
    stA1 = *(const uint4*)(ap + 8);
    if (BF32) {
      const float* bp = bSrcF + (size_t)k0 * ldb;
      stBf[0] = *(const float4*)(bp);
      stBf[1] = *(const float4*)(bp + 4);
      stBf[2] = *(const float4*)(bp + 8);
      stBf[3] = *(const float4*)(bp + 12);
    } else {
      const bf16_t* bp = bSrcH + (size_t)k0 * ldb;
      stB0 = *(const uint4*)bp;
      stB1 = *(const uint4*)(bp + 8);
    }
  };
  auto lstore = [&](int buf) {
    bf16_t* la = &ldsA[buf][arow * 32 + acol];
    ((uint4*)la)[0] = stA0;
    ((uint4*)la)[1] = stA1;
    bf16_t* lb = &ldsB[buf][brow * 128 + bcol];
    if (BF32) {
      union { uint4 q[2]; bf16_t s[16]; } cv;
#pragma unroll
      for (int j = 0; j < 4; ++j) {
        cv.s[j*4+0] = (bf16_t)stBf[j].x; cv.s[j*4+1] = (bf16_t)stBf[j].y;
        cv.s[j*4+2] = (bf16_t)stBf[j].z; cv.s[j*4+3] = (bf16_t)stBf[j].w;
      }
      ((uint4*)lb)[0] = cv.q[0];
      ((uint4*)lb)[1] = cv.q[1];
    } else {
      ((uint4*)lb)[0] = stB0;
      ((uint4*)lb)[1] = stB1;
    }
  };
  auto domma = [&](int buf) {
    Frag a[2], b[4];
#pragma unroll
    for (int mi = 0; mi < 2; ++mi) {
      const bf16_t* p = &ldsA[buf][(waveM + mi * 16 + l15) * 32 + sel];
      a[mi].q[0] = *(const uint4*)p;        // K = sel .. sel+7
      a[mi].q[1] = *(const uint4*)(p + 16); // K = sel+16 .. sel+23
    }
#pragma unroll
    for (int ni = 0; ni < 4; ++ni) {
      const bf16_t* p = &ldsB[buf][sel * 128 + waveN + ni * 16 + l15];
#pragma unroll
      for (int s = 0; s < 8; ++s) { b[ni].s[s] = p[s * 128]; b[ni].s[8 + s] = p[(16 + s) * 128]; }
    }
#pragma unroll
    for (int mi = 0; mi < 2; ++mi)
#pragma unroll
      for (int ni = 0; ni < 4; ++ni)
        acc[mi][ni] = wmma_bf16(a[mi].v, b[ni].v, acc[mi][ni]);
  };

  gload(0);
  lstore(0);
  __syncthreads();
  int cur = 0;
  for (int k0 = 32; k0 < Kd; k0 += 32) {
    // prefetch the B stream one more step ahead (dominant HBM/L2 traffic)
    if (k0 + 32 < Kd) {
      if (BF32) __builtin_prefetch(bSrcF + (size_t)(k0 + 32) * ldb, 0, 1);
      else      __builtin_prefetch(bSrcH + (size_t)(k0 + 32) * ldb, 0, 1);
    }
    gload(k0);        // global loads issue before math; WMMA hides latency
    domma(cur);       // math on current LDS buffer
    int nxt = cur ^ 1;
    lstore(nxt);      // drain staged registers into the other buffer
    __syncthreads();
    cur = nxt;
  }
  domma(cur);

#pragma unroll
  for (int mi = 0; mi < 2; ++mi)
#pragma unroll
    for (int ni = 0; ni < 4; ++ni)
#pragma unroll
      for (int g = 0; g < 8; ++g) {
        int row = blockM + waveM + mi * 16 + g + 8 * hi;
        int col = blockN + waveN + ni * 16 + l15;
        float vv = acc[mi][ni][g];
        if (GELU) vv = 0.5f * vv * (1.0f + erff(vv * 0.70710678118f));
        size_t idx = (size_t)row * ldc + col;
        if (OUTM == 2)      ((float*)C)[idx]  = vv + resid[idx];
        else if (OUTM == 1) ((bf16_t*)C)[idx] = (bf16_t)vv;
        else                ((float*)C)[idx]  = vv;
      }
}

// ---------- K transpose: qkv[b,t,D + h*64+d] -> Kt[b,h,d,t] ----------
__global__ __launch_bounds__(256) void ktrans_kernel(const bf16_t* __restrict__ qkv,
                                                     bf16_t* __restrict__ kt) {
  int idx = blockIdx.x * 256 + threadIdx.x;
  int t = idx & (T_ - 1);
  int d = (idx >> 10) & (HD_ - 1);
  int h = (idx >> 16) & (H_ - 1);
  int b = idx >> 20;
  kt[idx] = qkv[((size_t)(b * T_ + t)) * (3 * D_) + D_ + h * HD_ + d];
}

// ---------- flash attention: 1 wave per (b,h,16-row q tile) ----------
__global__ __launch_bounds__(32) void attn_kernel(const bf16_t* __restrict__ qkv,
                                                  const bf16_t* __restrict__ Kt,
                                                  bf16_t* __restrict__ aout) {
  int qt = blockIdx.x, h = blockIdx.y, b = blockIdx.z;
  int lane = threadIdx.x;
  int l15 = lane & 15, hi = lane >> 4, sel = hi * 8;
  int q0 = qt * 16;

  __shared__ __align__(16) bf16_t ptile[16 * 32];

  // preload Q fragments (d-chunks 0 and 32)
  Frag qa[2];
  {
    int qrow = q0 + l15;
    const bf16_t* qp = qkv + ((size_t)(b * T_ + qrow)) * (3 * D_) + h * HD_;
#pragma unroll
    for (int dc = 0; dc < 2; ++dc) {
      const bf16_t* p = qp + dc * 32 + sel;
      qa[dc].q[0] = *(const uint4*)p;
      qa[dc].q[1] = *(const uint4*)(p + 16);
    }
  }

  v8f o[4] = {};
  float mrow[8], lrow[8];
#pragma unroll
  for (int g = 0; g < 8; ++g) { mrow[g] = NEG_INF; lrow[g] = 0.0f; }

  int kend = q0 + 16; // keys 0 .. q0+15
  for (int kc = 0; kc < kend; kc += 32) {
    v8f sc[2] = {};
#pragma unroll
    for (int sub = 0; sub < 2; ++sub)
#pragma unroll
      for (int dc = 0; dc < 2; ++dc) {
        const bf16_t* kp = Kt + (((size_t)(b * H_ + h)) * HD_ + dc * 32 + sel) * T_ + kc + sub * 16 + l15;
        Frag kb;
#pragma unroll
        for (int s = 0; s < 8; ++s) { kb.s[s] = kp[(size_t)s * T_]; kb.s[8 + s] = kp[(size_t)(16 + s) * T_]; }
        sc[sub] = wmma_bf16(qa[dc].v, kb.v, sc[sub]);
      }

#pragma unroll
    for (int g = 0; g < 8; ++g) {
      int row = q0 + g + 8 * hi;
      float va = sc[0][g] * SCALE_; if (kc + l15 > row)      va = NEG_INF;
      float vb = sc[1][g] * SCALE_; if (kc + 16 + l15 > row) vb = NEG_INF;
      float mx = fmaxf(va, vb);
      mx = fmaxf(mx, __shfl_xor(mx, 1)); mx = fmaxf(mx, __shfl_xor(mx, 2));
      mx = fmaxf(mx, __shfl_xor(mx, 4)); mx = fmaxf(mx, __shfl_xor(mx, 8));
      float nm = fmaxf(mrow[g], mx);
      float scl = __expf(mrow[g] - nm);
      float pa = __expf(va - nm);
      float pb = __expf(vb - nm);
      float rsum = pa + pb;
      rsum += __shfl_xor(rsum, 1); rsum += __shfl_xor(rsum, 2);
      rsum += __shfl_xor(rsum, 4); rsum += __shfl_xor(rsum, 8);
      lrow[g] = lrow[g] * scl + rsum;
      mrow[g] = nm;
      o[0][g] *= scl; o[1][g] *= scl; o[2][g] *= scl; o[3][g] *= scl;
      int r = g + 8 * hi;
      ptile[r * 32 + l15]      = (bf16_t)pa;
      ptile[r * 32 + 16 + l15] = (bf16_t)pb;
    }
    __syncthreads();
    Frag pf;
    {
      const bf16_t* pp = ptile + l15 * 32 + sel;
      pf.q[0] = *(const uint4*)pp;
      pf.q[1] = *(const uint4*)(pp + 16);
    }
    __syncthreads();
    const bf16_t* vbase = qkv + ((size_t)(b * T_ + kc + sel)) * (3 * D_) + 2 * D_ + h * HD_ + l15;
#pragma unroll
    for (int nf = 0; nf < 4; ++nf) {
      Frag vf;
#pragma unroll
      for (int s = 0; s < 8; ++s) { vf.s[s] = vbase[(size_t)s * (3 * D_) + nf * 16]; vf.s[8 + s] = vbase[(size_t)(16 + s) * (3 * D_) + nf * 16]; }
      o[nf] = wmma_bf16(pf.v, vf.v, o[nf]);
    }
  }

#pragma unroll
  for (int nf = 0; nf < 4; ++nf)
#pragma unroll
    for (int g = 0; g < 8; ++g) {
      int row = q0 + g + 8 * hi;
      int col = h * HD_ + nf * 16 + l15;
      aout[((size_t)(b * T_ + row)) * D_ + col] = (bf16_t)(o[nf][g] / lrow[g]);
    }
}

// ---------- router: logits + top-2 + softmax gates (k-major layout) ----------
__global__ __launch_bounds__(256) void router_kernel(const float* __restrict__ xf,
                                                     const float* __restrict__ rw,
                                                     int* __restrict__ eidx,
                                                     float* __restrict__ gate) {
  int n = blockIdx.x, t = threadIdx.x;
  float acc[E_] = {};
  const float* xr = xf + (size_t)n * D_;
  for (int d = t; d < D_; d += 256) {
    float xv = xr[d];
#pragma unroll
    for (int e = 0; e < E_; ++e) acc[e] += xv * rw[e * D_ + d];
  }
  __shared__ float red[E_][256];
#pragma unroll
  for (int e = 0; e < E_; ++e) red[e][t] = acc[e];
  __syncthreads();
  for (int st = 128; st > 0; st >>= 1) {
    if (t < st)
#pragma unroll
      for (int e = 0; e < E_; ++e) red[e][t] += red[e][t + st];
    __syncthreads();
  }
  if (t == 0) {
    int i0 = 0; float v0 = red[0][0];
    for (int e = 1; e < E_; ++e) { float v = red[e][0]; if (v > v0) { v0 = v; i0 = e; } }
    int i1 = -1; float v1 = NEG_INF;
    for (int e = 0; e < E_; ++e) { if (e == i0) continue; float v = red[e][0]; if (v > v1) { v1 = v; i1 = e; } }
    float e1 = __expf(v1 - v0);
    float g0 = 1.0f / (1.0f + e1);
    float g1 = e1 / (1.0f + e1);
    eidx[0 * N_ + n] = i0; gate[0 * N_ + n] = g0;
    eidx[1 * N_ + n] = i1; gate[1 * N_ + n] = g1;
  }
}

// ---------- deterministic capacity ranks (k-major cumsum, hierarchical) ----------
__global__ __launch_bounds__(32) void scan1_kernel(const int* __restrict__ eidx,
                                                   int* __restrict__ chunkCount,
                                                   int* __restrict__ rankInChunk) {
  int c = blockIdx.x, lane = threadIdx.x;
  int i = c * 32 + lane;
  int e = eidx[i];
  int r = 0;
#pragma unroll
  for (int j = 0; j < E_; ++j) {
    unsigned mm = (unsigned)__ballot(e == j);
    if (e == j)   r = __popc(mm & ((1u << lane) - 1u));
    if (lane == j) chunkCount[c * E_ + j] = __popc(mm);
  }
  rankInChunk[i] = r;
}

__global__ void scan2_kernel(const int* __restrict__ chunkCount, int* __restrict__ chunkBase) {
  int e = threadIdx.x;
  if (e >= E_) return;
  int run = 0;
  for (int c = 0; c < (KTOP * N_) / 32; ++c) {
    chunkBase[c * E_ + e] = run;
    run += chunkCount[c * E_ + e];
  }
}

// ---------- dispatch: copy token rows into expert slots ----------
__global__ __launch_bounds__(128) void dispatch_kernel(const int* __restrict__ eidx,
                                                       const float* __restrict__ gate,
                                                       const int* __restrict__ chunkBase,
                                                       const int* __restrict__ rankInChunk,
                                                       const bf16_t* __restrict__ xb,
                                                       bf16_t* __restrict__ buf,
                                                       int* __restrict__ slotArr,
                                                       float* __restrict__ wtArr) {
  int i = blockIdx.x; // k-major assignment index
  int e = eidx[i];
  int rank = chunkBase[(i >> 5) * E_ + e] + rankInChunk[i];
  bool valid = rank < CAP_;
  int slot = e * CAP_ + (valid ? rank : (CAP_ - 1));
  if (threadIdx.x == 0) { slotArr[i] = slot; wtArr[i] = valid ? gate[i] : 0.0f; }
  if (!valid) return;
  int n = i & (N_ - 1);
  const uint4* src = (const uint4*)(xb + (size_t)n * D_);
  uint4* dst = (uint4*)(buf + (size_t)slot * D_);
  dst[threadIdx.x] = src[threadIdx.x]; // 128 threads x 16B = 2KB row
}

// ---------- combine: out += sum_k wt * expOut[slot] ----------
__global__ __launch_bounds__(256) void combine_kernel(const bf16_t* __restrict__ expOut,
                                                      const int* __restrict__ slotArr,
                                                      const float* __restrict__ wtArr,
                                                      float* __restrict__ out) {
  int n = blockIdx.x, t = threadIdx.x;
  float w0 = wtArr[n],       w1 = wtArr[N_ + n];
  int   s0 = slotArr[n],     s1 = slotArr[N_ + n];
  int d0 = t * 4;
  size_t ob = (size_t)n * D_ + d0;
#pragma unroll
  for (int j = 0; j < 4; ++j) {
    float acc = out[ob + j];
    if (w0 != 0.0f) acc += w0 * (float)expOut[(size_t)s0 * D_ + d0 + j];
    if (w1 != 0.0f) acc += w1 * (float)expOut[(size_t)s1 * D_ + d0 + j];
    out[ob + j] = acc;
  }
}

// ---------- host orchestration ----------
extern "C" void kernel_launch(void* const* d_in, const int* in_sizes, int n_in,
                              void* d_out, int out_size, void* d_ws, size_t ws_size,
                              hipStream_t stream) {
  const float* x       = (const float*)d_in[0];
  const float* ln1w    = (const float*)d_in[1];
  const float* ln1b    = (const float*)d_in[2];
  const float* ln2w    = (const float*)d_in[3];
  const float* ln2b    = (const float*)d_in[4];
  const float* qkvw    = (const float*)d_in[5];
  const float* projw   = (const float*)d_in[6];
  const float* routerw = (const float*)d_in[7];
  const float* w1      = (const float*)d_in[8];
  const float* w2      = (const float*)d_in[9];
  float* out = (float*)d_out;

  char* ws = (char*)d_ws;
  size_t off = 0;
  auto alloc = [&](size_t bytes) -> char* {
    char* p = ws + off;
    off += (bytes + 255) & ~(size_t)255;
    return p;
  };

  bf16_t* qkvwT  = (bf16_t*)alloc((size_t)D_ * 3 * D_ * 2);
  bf16_t* projwT = (bf16_t*)alloc((size_t)D_ * D_ * 2);
  bf16_t* lnb    = (bf16_t*)alloc((size_t)N_ * D_ * 2);
  float*  ln2f   = (float*) alloc((size_t)N_ * D_ * 4);
  bf16_t* qkvb   = (bf16_t*)alloc((size_t)N_ * 3 * D_ * 2);
  bf16_t* Kt     = (bf16_t*)alloc((size_t)B_ * H_ * HD_ * T_ * 2);
  bf16_t* attnb  = (bf16_t*)alloc((size_t)N_ * D_ * 2);
  bf16_t* buf    = (bf16_t*)alloc((size_t)E_ * CAP_ * D_ * 2);
  bf16_t* hbuf   = (bf16_t*)alloc((size_t)CAP_ * 4 * D_ * 2);
  bf16_t* expOut = (bf16_t*)alloc((size_t)E_ * CAP_ * D_ * 2);
  int*    eidx   = (int*)   alloc((size_t)KTOP * N_ * 4);
  float*  gate   = (float*) alloc((size_t)KTOP * N_ * 4);
  int*    rnk    = (int*)   alloc((size_t)KTOP * N_ * 4);
  int*    cc     = (int*)   alloc((size_t)(KTOP * N_ / 32) * E_ * 4);
  int*    cb     = (int*)   alloc((size_t)(KTOP * N_ / 32) * E_ * 4);
  int*    slotA  = (int*)   alloc((size_t)KTOP * N_ * 4);
  float*  wtA    = (float*) alloc((size_t)KTOP * N_ * 4);
  (void)ws_size; (void)in_sizes; (void)n_in; (void)out_size;

  // weight packing (bf16, transposed for GEMM B layout)
  transpack_kernel<<<dim3(D_/32, 3*D_/32), dim3(32,8), 0, stream>>>(qkvw, qkvwT, 3*D_, D_);
  transpack_kernel<<<dim3(D_/32, D_/32),   dim3(32,8), 0, stream>>>(projw, projwT, D_, D_);

  // attention path
  ln_kernel<<<N_, 256, 0, stream>>>(x, ln1w, ln1b, lnb, nullptr);
  gemm_wmma<0,1,0><<<dim3(3*D_/128, N_/128), dim3(32,8), 0, stream>>>(
      lnb, D_, qkvwT, 3*D_, qkvb, 3*D_, nullptr);
  ktrans_kernel<<<(B_*H_*HD_*T_)/256, 256, 0, stream>>>(qkvb, Kt);
  attn_kernel<<<dim3(T_/16, H_, B_), 32, 0, stream>>>(qkvb, Kt, attnb);
  gemm_wmma<0,2,0><<<dim3(D_/128, N_/128), dim3(32,8), 0, stream>>>(
      attnb, D_, projwT, D_, out, D_, /*resid=*/x);

  // MoE path
  ln_kernel<<<N_, 256, 0, stream>>>(out, ln2w, ln2b, lnb, ln2f);
  router_kernel<<<N_, 256, 0, stream>>>(ln2f, routerw, eidx, gate);
  scan1_kernel<<<(KTOP*N_)/32, 32, 0, stream>>>(eidx, cc, rnk);
  scan2_kernel<<<1, E_, 0, stream>>>(cc, cb);
  dispatch_kernel<<<KTOP*N_, 128, 0, stream>>>(eidx, gate, cb, rnk, lnb, buf, slotA, wtA);

  for (int e = 0; e < E_; ++e) {
    gemm_wmma<1,1,1><<<dim3(4*D_/128, CAP_/128), dim3(32,8), 0, stream>>>(
        buf + (size_t)e * CAP_ * D_, D_,
        w1 + (size_t)e * D_ * 4 * D_, 4*D_,
        hbuf, 4*D_, nullptr);
    gemm_wmma<1,1,0><<<dim3(D_/128, CAP_/128), dim3(32,8), 0, stream>>>(
        hbuf, 4*D_,
        w2 + (size_t)e * 4 * D_ * D_, D_,
        expOut + (size_t)e * CAP_ * D_, D_, nullptr);
  }

  combine_kernel<<<N_, 256, 0, stream>>>(expOut, slotA, wtA, out);
}
